// DGCNNEncoder_58162447123323
// MI455X (gfx1250) — compile-verified, matrix operand order checked by reading, not run on hardware
//
#include <hip/hip_runtime.h>

// ---------------------------------------------------------------------------
// DGCNN encoder for MI455X (gfx1250, wave32). 1x1-conv einsums run on
// v_wmma_f32_16x16x32_f16 with f16 A/B and f32 accumulate. Weights are
// pre-converted to a zero-padded f16 tile so the GEMM inner loop is pure
// b128 loads + WMMA (no guards / per-element converts). kNN / FPS / gathers
// are lean VALU kernels. All scratch lives in d_ws.
// ---------------------------------------------------------------------------

typedef __attribute__((ext_vector_type(16))) _Float16 v16h;
typedef __attribute__((ext_vector_type(8)))  _Float16 v8h;
typedef __attribute__((ext_vector_type(8)))  float    v8f;

#define BATCH   2
#define NPTS0   8192
#define NPTS1   2048
#define NPTS2   512
#define KNN     16
#define NGROUPS 8

// ---------------------------------------------------------------------------
// Stem: F0[b,o,n] = sum_c xyz[b,c,n]*W[o,c] + bias[o]; also copy P0 to output.
// ---------------------------------------------------------------------------
__global__ void stem_copy_kernel(const float* __restrict__ xyz,
                                 const float* __restrict__ sw,
                                 const float* __restrict__ sb,
                                 float* __restrict__ F0,
                                 float* __restrict__ P0out) {
  int n = blockIdx.x * blockDim.x + threadIdx.x;
  int b = blockIdx.y;
  if (n >= NPTS0) return;
  float x = xyz[(b * 3 + 0) * NPTS0 + n];
  float y = xyz[(b * 3 + 1) * NPTS0 + n];
  float z = xyz[(b * 3 + 2) * NPTS0 + n];
  P0out[(b * 3 + 0) * NPTS0 + n] = x;
  P0out[(b * 3 + 1) * NPTS0 + n] = y;
  P0out[(b * 3 + 2) * NPTS0 + n] = z;
#pragma unroll
  for (int o = 0; o < 8; ++o) {
    F0[(b * 8 + o) * NPTS0 + n] =
        sw[o * 3 + 0] * x + sw[o * 3 + 1] * y + sw[o * 3 + 2] * z + sb[o];
  }
}

// ---------------------------------------------------------------------------
// kNN (k=16): one query per thread, keys tiled through LDS, register
// insertion-sorted top-16 (constant-index bubble insert stays in VGPRs).
// ---------------------------------------------------------------------------
__global__ void knn_kernel(const float* __restrict__ Pq,
                           const float* __restrict__ Pk,
                           int Nq, int Np, int* __restrict__ idxOut) {
  __shared__ float sx[256], sy[256], sz[256];
  int b = blockIdx.y;
  int q = blockIdx.x * 256 + threadIdx.x;
  bool valid = q < Nq;
  float qx = 0.f, qy = 0.f, qz = 0.f;
  if (valid) {
    qx = Pq[(b * 3 + 0) * Nq + q];
    qy = Pq[(b * 3 + 1) * Nq + q];
    qz = Pq[(b * 3 + 2) * Nq + q];
  }
  float bd[KNN];
  int   bi[KNN];
#pragma unroll
  for (int u = 0; u < KNN; ++u) { bd[u] = 3.4e38f; bi[u] = 0; }

  for (int base = 0; base < Np; base += 256) {
    int j = base + threadIdx.x;
    if (j < Np) {
      sx[threadIdx.x] = Pk[(b * 3 + 0) * Np + j];
      sy[threadIdx.x] = Pk[(b * 3 + 1) * Np + j];
      sz[threadIdx.x] = Pk[(b * 3 + 2) * Np + j];
    }
    __syncthreads();
    int lim = min(256, Np - base);
    for (int t = 0; t < lim; ++t) {
      float dx = qx - sx[t], dy = qy - sy[t], dz = qz - sz[t];
      float d = dx * dx + dy * dy + dz * dz;
      if (d < bd[KNN - 1]) {
        float cd = d; int ci = base + t;
#pragma unroll
        for (int u = 0; u < KNN; ++u) {
          if (cd < bd[u]) {
            float td = bd[u]; int ti = bi[u];
            bd[u] = cd; bi[u] = ci;
            cd = td; ci = ti;
          }
        }
      }
    }
    __syncthreads();
  }
  if (valid) {
#pragma unroll
    for (int u = 0; u < KNN; ++u) idxOut[(b * Nq + q) * KNN + u] = bi[u];
  }
}

// ---------------------------------------------------------------------------
// Pair feature build: row r=(b,n,kk) -> [nbr-Fi | Fi | 0-pad] as f16, Kpad cols
// ---------------------------------------------------------------------------
__global__ void pair_build_kernel(const float* __restrict__ Fq,
                                  const float* __restrict__ Fk,
                                  const int* __restrict__ idx,
                                  _Float16* __restrict__ pair,
                                  int C, int Kpad, int Nq, int Np) {
  int r = blockIdx.x * blockDim.x + threadIdx.x;
  int total = BATCH * Nq * KNN;
  if (r >= total) return;
  int n = (r / KNN) % Nq;
  int b = r / (KNN * Nq);
  int j = idx[r];
  _Float16* row = pair + (size_t)r * Kpad;
  for (int c = 0; c < C; ++c) {
    float fi = Fq[(b * C + c) * Nq + n];
    float fn = Fk[(b * C + c) * Np + j];
    row[c]     = (_Float16)(fn - fi);
    row[C + c] = (_Float16)fi;
  }
  for (int c = 2 * C; c < Kpad; ++c) row[c] = (_Float16)0.f;
}

// ---------------------------------------------------------------------------
// Weight pre-convert: W (Co x Kin, f32) -> Wh (Co x Kpad, f16, zero-padded).
// ---------------------------------------------------------------------------
__global__ void convert_w_kernel(const float* __restrict__ W,
                                 _Float16* __restrict__ Wh,
                                 int Co, int Kin, int Kpad) {
  int i = blockIdx.x * blockDim.x + threadIdx.x;
  int total = Co * Kpad;
  if (i >= total) return;
  int k = i % Kpad;
  int o = i / Kpad;
  Wh[i] = (k < Kin) ? (_Float16)W[o * Kin + k] : (_Float16)0.f;
}

// ---------------------------------------------------------------------------
// WMMA GEMM: O[m,o] = sum_k pair[m,k] * Wh[o,k]. One wave per 16-row M tile;
// the wave computes ALL Co/16 N tiles, reusing the A fragment (accumulators
// stay resident: up to 8 x v8f = 64 VGPRs at wave32).
// A/B layout (16-bit, 16x32): lane holds row/col (l&15); K packs as two
// contiguous 8-half chunks at offsets 8g and 16+8g (g=l>>4) -> b128 loads.
// D layout: lane = column (l&15), VGPR j = row j+8g.
// ---------------------------------------------------------------------------
template <int KPAD, int CO>
__global__ void wmma_gemm_kernel(const _Float16* __restrict__ pair,
                                 const _Float16* __restrict__ Wh,
                                 float* __restrict__ preact, int M) {
  constexpr int TN = CO / 16;
  int wave = (blockIdx.x * blockDim.x + threadIdx.x) >> 5;
  int lane = threadIdx.x & 31;
  if (wave >= (M >> 4)) return;
  int g  = lane >> 4;
  int lr = lane & 15;
  const _Float16* rowp = pair + (size_t)(wave * 16 + lr) * KPAD;

  v8f acc[TN];
#pragma unroll
  for (int t = 0; t < TN; ++t) acc[t] = (v8f){0.f, 0.f, 0.f, 0.f, 0.f, 0.f, 0.f, 0.f};

#pragma unroll
  for (int k0 = 0; k0 < KPAD; k0 += 32) {
    v8h alo = *(const v8h*)(rowp + k0 + 8 * g);
    v8h ahi = *(const v8h*)(rowp + k0 + 16 + 8 * g);
    v16h a = __builtin_shufflevector(alo, ahi, 0, 1, 2, 3, 4, 5, 6, 7,
                                     8, 9, 10, 11, 12, 13, 14, 15);
#pragma unroll
    for (int t = 0; t < TN; ++t) {
      const _Float16* wp = Wh + (size_t)(t * 16 + lr) * KPAD;
      v8h blo = *(const v8h*)(wp + k0 + 8 * g);
      v8h bhi = *(const v8h*)(wp + k0 + 16 + 8 * g);
      v16h bm = __builtin_shufflevector(blo, bhi, 0, 1, 2, 3, 4, 5, 6, 7,
                                        8, 9, 10, 11, 12, 13, 14, 15);
      acc[t] = __builtin_amdgcn_wmma_f32_16x16x32_f16(
          /*neg_a=*/false, a, /*neg_b=*/false, bm,
          /*c_mod=*/(short)0, acc[t], /*reuse_a=*/false, /*reuse_b=*/false);
    }
  }

#pragma unroll
  for (int t = 0; t < TN; ++t) {
#pragma unroll
    for (int j = 0; j < 8; ++j) {
      int m = wave * 16 + j + 8 * g;
      preact[(size_t)m * CO + t * 16 + lr] = acc[t][j];
    }
  }
}

// ---------------------------------------------------------------------------
// GroupNorm pass 1: per (batch, group) sum / sumsq over (C/G, N, K).
// ---------------------------------------------------------------------------
__global__ void zero_kernel(float* __restrict__ p, int n) {
  int i = blockIdx.x * blockDim.x + threadIdx.x;
  if (i < n) p[i] = 0.f;
}

__global__ void stats_kernel(const float* __restrict__ preact,
                             float* __restrict__ stats, int Mb, int Co) {
  int b = blockIdx.y, grp = blockIdx.z;
  int cpg = Co / NGROUPS;
  size_t total = (size_t)Mb * cpg;
  float s = 0.f, sq = 0.f;
  for (size_t i = blockIdx.x * (size_t)blockDim.x + threadIdx.x; i < total;
       i += (size_t)gridDim.x * blockDim.x) {
    int o = grp * cpg + (int)(i % cpg);
    size_t m = i / cpg;
    float v = preact[((size_t)b * Mb + m) * Co + o];
    s += v; sq += v * v;
  }
  __shared__ float ss[256], ssq[256];
  ss[threadIdx.x] = s; ssq[threadIdx.x] = sq;
  __syncthreads();
  for (int off = 128; off > 0; off >>= 1) {
    if ((int)threadIdx.x < off) {
      ss[threadIdx.x]  += ss[threadIdx.x + off];
      ssq[threadIdx.x] += ssq[threadIdx.x + off];
    }
    __syncthreads();
  }
  if (threadIdx.x == 0) {
    atomicAdd(&stats[(b * NGROUPS + grp) * 2 + 0], ss[0]);
    atomicAdd(&stats[(b * NGROUPS + grp) * 2 + 1], ssq[0]);
  }
}

// ---------------------------------------------------------------------------
// GroupNorm pass 2 + ReLU + max over k neighbors -> F[b,o,n].
// ---------------------------------------------------------------------------
__global__ void norm_max_kernel(const float* __restrict__ preact,
                                const float* __restrict__ stats,
                                const float* __restrict__ gamma,
                                const float* __restrict__ beta,
                                float* __restrict__ Fout, int Nq, int Co) {
  int i = blockIdx.x * blockDim.x + threadIdx.x;
  int total = BATCH * Co * Nq;
  if (i >= total) return;
  int n = i % Nq;
  int o = (i / Nq) % Co;
  int b = i / (Nq * Co);
  int cpg = Co / NGROUPS;
  int grp = o / cpg;
  float cnt = (float)cpg * (float)Nq * (float)KNN;
  float s  = stats[(b * NGROUPS + grp) * 2 + 0];
  float sq = stats[(b * NGROUPS + grp) * 2 + 1];
  float mu = s / cnt;
  float var = fmaxf(sq / cnt - mu * mu, 0.f);
  float inv = rsqrtf(var + 1e-5f);
  float ga = gamma[o], be = beta[o];
  const float* base = preact + ((size_t)(b * Nq + n) * KNN) * Co + o;
  float mx = -3.4e38f;
#pragma unroll
  for (int kk = 0; kk < KNN; ++kk) {
    float v = base[(size_t)kk * Co];
    v = fmaf(ga * (v - mu), inv, be);
    v = fmaxf(v, 0.f);
    mx = fmaxf(mx, v);
  }
  Fout[(b * Co + o) * Nq + n] = mx;
}

// ---------------------------------------------------------------------------
// Farthest point sampling: one block per batch, distances live in LDS.
// Serial over m-1 steps (true dependency); 1024-thread argmax tree reduce.
// ---------------------------------------------------------------------------
__global__ __launch_bounds__(1024)
void fps_kernel(const float* __restrict__ P, int N, int m,
                int* __restrict__ idxOut) {
  int b = blockIdx.x;
  const float* px = P + (size_t)(b * 3 + 0) * N;
  const float* py = P + (size_t)(b * 3 + 1) * N;
  const float* pz = P + (size_t)(b * 3 + 2) * N;
  __shared__ float dists[NPTS0];
  __shared__ float rv[1024];
  __shared__ int   ri[1024];
  int tid = threadIdx.x;
  for (int i = tid; i < N; i += 1024) dists[i] = 1e10f;
  if (tid == 0) idxOut[b * m + 0] = 0;
  __syncthreads();
  int last = 0;
  for (int step = 1; step < m; ++step) {
    float lx = px[last], ly = py[last], lz = pz[last];
    float bestv = -1.f; int besti = 0;
    for (int i = tid; i < N; i += 1024) {
      float dx = px[i] - lx, dy = py[i] - ly, dz = pz[i] - lz;
      float d = dx * dx + dy * dy + dz * dz;
      float dm = fminf(dists[i], d);
      dists[i] = dm;
      if (dm > bestv) { bestv = dm; besti = i; }
    }
    rv[tid] = bestv; ri[tid] = besti;
    __syncthreads();
    for (int off = 512; off > 0; off >>= 1) {
      if (tid < off) {
        bool take = (rv[tid + off] > rv[tid]) ||
                    (rv[tid + off] == rv[tid] && ri[tid + off] < ri[tid]);
        if (take) { rv[tid] = rv[tid + off]; ri[tid] = ri[tid + off]; }
      }
      __syncthreads();
    }
    last = ri[0];
    if (tid == 0) idxOut[b * m + step] = last;
    __syncthreads();
  }
}

// ---------------------------------------------------------------------------
// Gather (B,C,N) -> (B,C,M) by per-batch index list.
// ---------------------------------------------------------------------------
__global__ void gather_kernel(const float* __restrict__ x,
                              const int* __restrict__ idx,
                              float* __restrict__ out, int C, int N, int M) {
  int i = blockIdx.x * blockDim.x + threadIdx.x;
  int total = BATCH * C * M;
  if (i >= total) return;
  int mm = i % M;
  int c = (i / M) % C;
  int b = i / (M * C);
  out[i] = x[(b * C + c) * N + idx[b * M + mm]];
}

// ---------------------------------------------------------------------------
// Host-side edge-conv pipeline.
// ---------------------------------------------------------------------------
static void run_edge_conv(const float* Fq, const float* Fk,
                          const float* Pq, const float* Pk,
                          const float* W, const float* gamma, const float* beta,
                          int C, int Co, int Nq, int Np,
                          int* idx_ws, _Float16* pair_ws, _Float16* wh_ws,
                          float* preact_ws, float* stats_ws, float* dest,
                          hipStream_t stream, bool doKnn) {
  int Kin = 2 * C;
  int Kpad = (Kin + 31) & ~31;
  if (doKnn) {
    knn_kernel<<<dim3((Nq + 255) / 256, BATCH), 256, 0, stream>>>(
        Pq, Pk, Nq, Np, idx_ws);
  }
  int rows = BATCH * Nq * KNN;
  pair_build_kernel<<<(rows + 255) / 256, 256, 0, stream>>>(
      Fq, Fk, idx_ws, pair_ws, C, Kpad, Nq, Np);
  convert_w_kernel<<<(Co * Kpad + 255) / 256, 256, 0, stream>>>(
      W, wh_ws, Co, Kin, Kpad);
  zero_kernel<<<1, 64, 0, stream>>>(stats_ws, BATCH * NGROUPS * 2);

  int waves = rows / 16;
  int blocks = (waves * 32 + 255) / 256;
  if (Kpad == 32 && Co == 32) {
    wmma_gemm_kernel<32, 32><<<blocks, 256, 0, stream>>>(pair_ws, wh_ws, preact_ws, rows);
  } else if (Kpad == 64 && Co == 64) {
    wmma_gemm_kernel<64, 64><<<blocks, 256, 0, stream>>>(pair_ws, wh_ws, preact_ws, rows);
  } else if (Kpad == 128 && Co == 64) {
    wmma_gemm_kernel<128, 64><<<blocks, 256, 0, stream>>>(pair_ws, wh_ws, preact_ws, rows);
  } else { // Kpad == 128 && Co == 128
    wmma_gemm_kernel<128, 128><<<blocks, 256, 0, stream>>>(pair_ws, wh_ws, preact_ws, rows);
  }

  stats_kernel<<<dim3(64, BATCH, NGROUPS), 256, 0, stream>>>(
      preact_ws, stats_ws, Nq * KNN, Co);
  int tot = BATCH * Co * Nq;
  norm_max_kernel<<<(tot + 255) / 256, 256, 0, stream>>>(
      preact_ws, stats_ws, gamma, beta, dest, Nq, Co);
}

extern "C" void kernel_launch(void* const* d_in, const int* in_sizes, int n_in,
                              void* d_out, int out_size, void* d_ws,
                              size_t ws_size, hipStream_t stream) {
  (void)in_sizes; (void)n_in; (void)out_size; (void)ws_size;
  const float* xyz    = (const float*)d_in[0];
  const float* stem_w = (const float*)d_in[1];
  const float* stem_b = (const float*)d_in[2];
  const float* w1 = (const float*)d_in[3];
  const float* g1 = (const float*)d_in[4];
  const float* b1 = (const float*)d_in[5];
  const float* w2 = (const float*)d_in[6];
  const float* g2 = (const float*)d_in[7];
  const float* b2 = (const float*)d_in[8];
  const float* w3 = (const float*)d_in[9];
  const float* g3 = (const float*)d_in[10];
  const float* b3 = (const float*)d_in[11];
  const float* w4 = (const float*)d_in[12];
  const float* g4 = (const float*)d_in[13];
  const float* b4 = (const float*)d_in[14];

  // ---- output slices (tuple flattened in return order) ----
  float* out   = (float*)d_out;
  float* oP0   = out;                            // (2,3,8192)
  float* oF0a  = oP0  + BATCH * 3   * NPTS0;     // (2,32,8192)
  float* oP1   = oF0a + BATCH * 32  * NPTS0;     // (2,3,2048)
  float* oF1a  = oP1  + BATCH * 3   * NPTS1;     // (2,64,2048)
  int*   oI1   = (int*)(oF1a + BATCH * 64 * NPTS1);  // (2,2048) int bits
  float* oP2   = (float*)(oI1 + BATCH * NPTS1);  // (2,3,512)
  float* oF2a  = oP2  + BATCH * 3   * NPTS2;     // (2,128,512)
  int*   oI2   = (int*)(oF2a + BATCH * 128 * NPTS2); // (2,512) int bits

  // ---- workspace carve-out ----
  char* w = (char*)d_ws;
  float* F0 = (float*)w;        w += (size_t)BATCH * 8 * NPTS0 * 4;
  int* idx_ws = (int*)w;        w += (size_t)BATCH * NPTS0 * KNN * 4;
  _Float16* pair_ws = (_Float16*)w;
                                w += (size_t)BATCH * NPTS0 * KNN * 32 * 2; // 16MB (max, conv1)
  _Float16* wh_ws = (_Float16*)w;
                                w += (size_t)128 * 128 * 2;                // 32KB (max Co*Kpad)
  float* preact_ws = (float*)w; w += (size_t)BATCH * NPTS0 * KNN * 32 * 4; // 33.5MB (max, conv1)
  float* stats_ws = (float*)w;  w += (size_t)BATCH * NGROUPS * 2 * 4;
  float* F1sk = (float*)w;      w += (size_t)BATCH * 32 * NPTS1 * 4;
  float* F2sk = (float*)w;      w += (size_t)BATCH * 64 * NPTS2 * 4;
  float* F2mid = (float*)w;     w += (size_t)BATCH * 64 * NPTS2 * 4;

  // ---- stage 0: stem + P0 copy ----
  stem_copy_kernel<<<dim3((NPTS0 + 255) / 256, BATCH), 256, 0, stream>>>(
      xyz, stem_w, stem_b, F0, oP0);

  // ---- edge conv 1: (F0,F0,P0,P0) -> F0a (2,32,8192) ----
  run_edge_conv(F0, F0, xyz, xyz, w1, g1, b1, 8, 32, NPTS0, NPTS0,
                idx_ws, pair_ws, wh_ws, preact_ws, stats_ws, oF0a, stream, true);

  // ---- FPS 1 + gathers ----
  fps_kernel<<<BATCH, 1024, 0, stream>>>(xyz, NPTS0, NPTS1, oI1);
  {
    int t = BATCH * 3 * NPTS1;
    gather_kernel<<<(t + 255) / 256, 256, 0, stream>>>(xyz, oI1, oP1, 3, NPTS0, NPTS1);
    t = BATCH * 32 * NPTS1;
    gather_kernel<<<(t + 255) / 256, 256, 0, stream>>>(oF0a, oI1, F1sk, 32, NPTS0, NPTS1);
  }

  // ---- edge conv 2: (F1sk, F0a, P1, P0) -> F1a (2,64,2048) ----
  run_edge_conv(F1sk, oF0a, oP1, xyz, w2, g2, b2, 32, 64, NPTS1, NPTS0,
                idx_ws, pair_ws, wh_ws, preact_ws, stats_ws, oF1a, stream, true);

  // ---- FPS 2 + gathers ----
  fps_kernel<<<BATCH, 1024, 0, stream>>>(oP1, NPTS1, NPTS2, oI2);
  {
    int t = BATCH * 3 * NPTS2;
    gather_kernel<<<(t + 255) / 256, 256, 0, stream>>>(oP1, oI2, oP2, 3, NPTS1, NPTS2);
    t = BATCH * 64 * NPTS2;
    gather_kernel<<<(t + 255) / 256, 256, 0, stream>>>(oF1a, oI2, F2sk, 64, NPTS1, NPTS2);
  }

  // ---- edge conv 3: (F2sk, F1a, P2, P1) -> F2mid (2,64,512) ----
  run_edge_conv(F2sk, oF1a, oP2, oP1, w3, g3, b3, 64, 64, NPTS2, NPTS1,
                idx_ws, pair_ws, wh_ws, preact_ws, stats_ws, F2mid, stream, true);

  // ---- edge conv 4: same kNN (same P2/P1) -> reuse idx_ws; F2a (2,128,512) ----
  run_edge_conv(F2mid, oF1a, oP2, oP1, w4, g4, b4, 64, 128, NPTS2, NPTS1,
                idx_ws, pair_ws, wh_ws, preact_ws, stats_ws, oF2a, stream, false);
}
